// MOTCAT_Surv_75565654606323
// MI455X (gfx1250) — matrix operand order repaired and checked by reading, not compiled, not run on hardware
//
#include <hip/hip_runtime.h>
#include <hip/hip_bf16.h>

// ---------------- types ----------------
typedef __attribute__((ext_vector_type(2)))  float  v2f;
typedef __attribute__((ext_vector_type(8)))  float  v8f;
typedef __attribute__((ext_vector_type(16))) __bf16 v16bf;

#define N_PATCH 8192
#define DIN     1024
#define DH      256
#define TOPK    6
#define SCALE   0.0625f   // 256^-0.5

// ---------------- helpers ----------------
__device__ __forceinline__ float leaky(float v) { return v > 0.f ? v : 0.01f * v; }

__device__ __forceinline__ float wred_sum(float v) {
#pragma unroll
  for (int o = 16; o > 0; o >>= 1) v += __shfl_xor(v, o, 32);
  return v;
}

__device__ __forceinline__ __bf16 f2bf(float f) {
  unsigned u = __float_as_uint(f);
  u += 0x7FFFu + ((u >> 16) & 1u);            // round-to-nearest-even
  union { unsigned short u16; __bf16 b; } cv;
  cv.u16 = (unsigned short)(u >> 16);
  return cv.b;
}

__device__ __forceinline__ void topk_insert(float v, int j, float* tv, int* ti) {
  if (!(v > tv[TOPK - 1])) return;
  tv[TOPK - 1] = v; ti[TOPK - 1] = j;
#pragma unroll
  for (int q = TOPK - 1; q > 0; --q) {
    if (tv[q] > tv[q - 1]) {
      float fv = tv[q]; tv[q] = tv[q - 1]; tv[q - 1] = fv;
      int   fi = ti[q]; ti[q] = ti[q - 1]; ti[q - 1] = fi;
    }
  }
}

// ---------------- fp32 WMMA GEMM:  C = act(A[MxK] @ W[KxN] + b) ----------------
// one wave per 64x16 output strip (4 M-subtiles share each strided B fragment;
// 4 independent accumulator chains keep the matrix pipe busy)
__global__ __launch_bounds__(128) void gemm_f32_kernel(
    const float* __restrict__ A, int lda,
    const float* __restrict__ W, int ldw,
    const float* __restrict__ bias,
    float* __restrict__ C, int ldc,
    int Nc, int Kd, int act, int accum) {
  const int tid  = threadIdx.x;
  const int wv   = tid >> 5;
  const int lane = tid & 31;
  const int half = lane >> 4;
  const int l15  = lane & 15;

  const int tiles_n = Nc >> 4;
  const int wave_id = blockIdx.x * 4 + wv;
  const int row0    = (wave_id / tiles_n) * 64;
  const int col0    = (wave_id % tiles_n) * 16;

  v8f c0 = {}, c1 = {}, c2 = {}, c3 = {};
  const float* A0 = A + (size_t)(row0 + l15) * lda + 2 * half;   // A[row][k+2*half]
  const float* A1 = A0 + (size_t)16 * lda;
  const float* A2 = A0 + (size_t)32 * lda;
  const float* A3 = A0 + (size_t)48 * lda;
  const float* Wc = W + (size_t)(2 * half) * ldw + col0 + l15;   // W[k+2*half][col]

  for (int k = 0; k < Kd; k += 4) {
    v2f b;  b.x  = Wc[(size_t)k * ldw];  b.y  = Wc[(size_t)(k + 1) * ldw];
    v2f a0; a0.x = A0[k];                a0.y = A0[k + 1];
    v2f a1; a1.x = A1[k];                a1.y = A1[k + 1];
    v2f a2; a2.x = A2[k];                a2.y = A2[k + 1];
    v2f a3; a3.x = A3[k];                a3.y = A3[k + 1];
    c0 = __builtin_amdgcn_wmma_f32_16x16x4_f32(false, a0, false, b, (short)0, c0, false, false);
    c1 = __builtin_amdgcn_wmma_f32_16x16x4_f32(false, a1, false, b, (short)0, c1, false, false);
    c2 = __builtin_amdgcn_wmma_f32_16x16x4_f32(false, a2, false, b, (short)0, c2, false, false);
    c3 = __builtin_amdgcn_wmma_f32_16x16x4_f32(false, a3, false, b, (short)0, c3, false, false);
  }

  const int   col = col0 + l15;
  const float bv  = bias[col];
#pragma unroll
  for (int s = 0; s < 4; ++s) {
    const v8f cs = (s == 0) ? c0 : (s == 1) ? c1 : (s == 2) ? c2 : c3;
#pragma unroll
    for (int r = 0; r < 8; ++r) {
      const int row = row0 + s * 16 + r + 8 * half;
      float v = cs[r] + bv;
      if (act) v = leaky(v);
      float* o = C + (size_t)row * ldc + col;
      if (accum) v += *o;
      *o = v;
    }
  }
}

// ---------------- column-sum partials over x_pre (for the mean mix) ----------------
__global__ __launch_bounds__(256) void colsum_partial_kernel(const float* __restrict__ x,
                                                             float* __restrict__ part) {
  const int c = threadIdx.x;            // 0..255
  const int b = blockIdx.x;             // 0..31  (256 rows each)
  float s = 0.f;
  const float* p = x + (size_t)(b << 8) * DH + c;
  for (int r = 0; r < 256; ++r) s += p[(size_t)r * DH];
  part[b * DH + c] = s;
}

__global__ __launch_bounds__(256) void mean_final_kernel(const float* __restrict__ part,
                                                         float* __restrict__ mean) {
  const int c = threadIdx.x;
  float s = 0.f;
#pragma unroll
  for (int b = 0; b < 32; ++b) s += part[b * DH + c];
  mean[c] = s * (1.0f / (float)N_PATCH);
}

__global__ __launch_bounds__(256) void mix_kernel(float* __restrict__ x,
                                                  const float* __restrict__ mean) {
  const size_t i = (size_t)blockIdx.x * 256 + threadIdx.x;
  x[i] = (x[i] + mean[i & (DH - 1)]) * 0.5f;
}

// ---------------- fp32 -> bf16 conversion of e_h*SCALE and e_t ----------------
__global__ __launch_bounds__(256) void cvt_kernel(const float* __restrict__ eh,
                                                  const float* __restrict__ et,
                                                  __bf16* __restrict__ ehb,
                                                  __bf16* __restrict__ etb) {
  const size_t i = (size_t)blockIdx.x * 256 + threadIdx.x;
  ehb[i] = f2bf(eh[i] * SCALE);
  etb[i] = f2bf(et[i]);
}

// ---------------- attention logits (bf16 WMMA) + per-row top-6 ----------------
// one block of 8 waves per 16-row strip; waves split 512 column tiles
__global__ __launch_bounds__(256) void attn_topk_kernel(const __bf16* __restrict__ ehb,
                                                        const __bf16* __restrict__ etb,
                                                        int* __restrict__ tidx) {
  const int tid  = threadIdx.x;
  const int w    = tid >> 5;
  const int lane = tid & 31;
  const int half = lane >> 4;
  const int l15  = lane & 15;
  const int row0 = blockIdx.x << 4;

  __shared__ float sc[8][16][17];
  __shared__ float stv[8][16][TOPK];
  __shared__ int   sti[8][16][TOPK];

  // hoist A fragments (the same 16 rows for all column tiles): 8 k-steps of 32
  v16bf af[8];
  const __bf16* Ar = ehb + (size_t)(row0 + l15) * DH;
#pragma unroll
  for (int kk = 0; kk < 8; ++kk) {
    const int kb0 = kk * 32 + (half ? 16 : 0);
#pragma unroll
    for (int v = 0; v < 8; ++v) {
      const int kb = kb0 + 2 * v;
      af[kk][2 * v]     = Ar[kb];
      af[kk][2 * v + 1] = Ar[kb + 1];
    }
  }

  float tv[TOPK]; int ti[TOPK];
#pragma unroll
  for (int k = 0; k < TOPK; ++k) { tv[k] = -3.0e38f; ti[k] = 0; }

  for (int jt = w; jt < (N_PATCH >> 4); jt += 8) {
    const int col0 = jt << 4;
    v8f c = {};
    const __bf16* Br = etb + (size_t)(col0 + l15) * DH + half * 16;
#pragma unroll
    for (int kk = 0; kk < 8; ++kk) {
      v16bf b = *(const v16bf*)(Br + kk * 32);
      c = __builtin_amdgcn_wmma_f32_16x16x32_bf16(false, af[kk], false, b, (short)0, c,
                                                  false, false);
    }
#pragma unroll
    for (int r = 0; r < 8; ++r) sc[w][r + 8 * half][l15] = c[r];
    // same-wave LDS RAW across lanes: wait for DS writes, wave is lockstep
    asm volatile("s_wait_dscnt 0x0" ::: "memory");
    if (lane < 16) {
#pragma unroll
      for (int cc = 0; cc < 16; ++cc) topk_insert(sc[w][l15][cc], col0 + cc, tv, ti);
    }
  }

  if (lane < 16) {
#pragma unroll
    for (int k = 0; k < TOPK; ++k) { stv[w][l15][k] = tv[k]; sti[w][l15][k] = ti[k]; }
  }
  __syncthreads();
  if (w == 0 && lane < 16) {
    float fv[TOPK]; int fi[TOPK];
#pragma unroll
    for (int k = 0; k < TOPK; ++k) { fv[k] = -3.0e38f; fi[k] = 0; }
#pragma unroll
    for (int ww = 0; ww < 8; ++ww)
#pragma unroll
      for (int k = 0; k < TOPK; ++k) topk_insert(stv[ww][l15][k], sti[ww][l15][k], fv, fi);
#pragma unroll
    for (int k = 0; k < TOPK; ++k) tidx[(row0 + l15) * TOPK + k] = fi[k];
  }
}

// ---------------- message construction: recompute fp32 logits, softmaxes, write t1/t2 ----------------
// one wave per row; lane owns 8 consecutive feature dims
__global__ __launch_bounds__(256) void message_kernel(const float* __restrict__ eh,
                                                      const float* __restrict__ et,
                                                      const int* __restrict__ tidx,
                                                      float* __restrict__ t1,
                                                      float* __restrict__ t2) {
  const int row  = blockIdx.x * 8 + (threadIdx.x >> 5);
  const int lane = threadIdx.x & 31;
  const int dbase = lane * 8;

  const float* ehr = eh + (size_t)row * DH + dbase;
  float ehv[8];
#pragma unroll
  for (int q = 0; q < 8; ++q) ehv[q] = ehr[q];

  int idx[TOPK];
#pragma unroll
  for (int k = 0; k < TOPK; ++k) idx[k] = tidx[row * TOPK + k];

  float nb[TOPK][8];
#pragma unroll
  for (int k = 0; k < TOPK; ++k) {
    const float* nr = et + (size_t)idx[k] * DH + dbase;
#pragma unroll
    for (int q = 0; q < 8; ++q) nb[k][q] = nr[q];
  }

  // exact fp32 logits + neighbor feature sums
  float lg[TOPK], nbs[TOPK];
#pragma unroll
  for (int k = 0; k < TOPK; ++k) {
    float d = 0.f, s = 0.f;
#pragma unroll
    for (int q = 0; q < 8; ++q) { d += ehv[q] * nb[k][q]; s += nb[k][q]; }
    lg[k]  = wred_sum(d) * SCALE;
    nbs[k] = wred_sum(s);
  }

  // p = softmax(lg)
  float m = lg[0];
#pragma unroll
  for (int k = 1; k < TOPK; ++k) m = fmaxf(m, lg[k]);
  float p[TOPK], ps = 0.f;
#pragma unroll
  for (int k = 0; k < TOPK; ++k) { p[k] = __expf(lg[k] - m); ps += p[k]; }
#pragma unroll
  for (int k = 0; k < TOPK; ++k) p[k] /= ps;

  // gate = tanh((2-p)*e_h + p*Nb);  ka_w = nbsum * gate_sum
  float ka[TOPK];
#pragma unroll
  for (int k = 0; k < TOPK; ++k) {
    float g = 0.f;
#pragma unroll
    for (int q = 0; q < 8; ++q) g += tanhf((2.f - p[k]) * ehv[q] + p[k] * nb[k][q]);
    ka[k] = nbs[k] * wred_sum(g);
  }

  // ka_p = softmax(ka)
  float km = ka[0];
#pragma unroll
  for (int k = 1; k < TOPK; ++k) km = fmaxf(km, ka[k]);
  float kp[TOPK], ks = 0.f;
#pragma unroll
  for (int k = 0; k < TOPK; ++k) { kp[k] = __expf(ka[k] - km); ks += kp[k]; }
#pragma unroll
  for (int k = 0; k < TOPK; ++k) kp[k] /= ks;

  // e_Nh, then t1 = e_h + e_Nh, t2 = e_h * e_Nh
  float* o1 = t1 + (size_t)row * DH + dbase;
  float* o2 = t2 + (size_t)row * DH + dbase;
#pragma unroll
  for (int q = 0; q < 8; ++q) {
    float acc = 0.f;
#pragma unroll
    for (int k = 0; k < TOPK; ++k) acc += kp[k] * nb[k][q];
    o1[q] = ehv[q] + acc;
    o2[q] = ehv[q] * acc;
  }
}

// ---------------- readout: g logits (wave per row) ----------------
__global__ __launch_bounds__(256) void glog_kernel(const float* __restrict__ h,
                                                   const float* __restrict__ Ag2,
                                                   const float* __restrict__ bg2,
                                                   float* __restrict__ gl) {
  const int row  = blockIdx.x * 8 + (threadIdx.x >> 5);
  const int lane = threadIdx.x & 31;
  const float* hr = h + (size_t)row * 128 + lane * 4;
  const float* ar = Ag2 + lane * 4;
  float d = hr[0] * ar[0] + hr[1] * ar[1] + hr[2] * ar[2] + hr[3] * ar[3];
  d = wred_sum(d);
  if (lane == 0) gl[row] = d + bg2[0];
}

__global__ __launch_bounds__(256) void softmax_red_kernel(const float* __restrict__ gl,
                                                          float* __restrict__ red) {
  __shared__ float s[256];
  const int t = threadIdx.x;
  float m = -3.0e38f;
  for (int i = t; i < N_PATCH; i += 256) m = fmaxf(m, gl[i]);
  s[t] = m; __syncthreads();
  for (int o = 128; o > 0; o >>= 1) { if (t < o) s[t] = fmaxf(s[t], s[t + o]); __syncthreads(); }
  const float mx = s[0]; __syncthreads();
  float sum = 0.f;
  for (int i = t; i < N_PATCH; i += 256) sum += __expf(gl[i] - mx);
  s[t] = sum; __syncthreads();
  for (int o = 128; o > 0; o >>= 1) { if (t < o) s[t] += s[t + o]; __syncthreads(); }
  if (t == 0) { red[0] = mx; red[1] = s[0]; }
}

// 32 deterministic row-partials of e_g, then a fixed-order combine
__global__ __launch_bounds__(256) void readout_partial_kernel(const float* __restrict__ gl,
                                                              const float* __restrict__ red,
                                                              const float* __restrict__ emsg,
                                                              float* __restrict__ part) {
  const int c = threadIdx.x;
  const int b = blockIdx.x;                 // 0..31, 256 rows each
  const float mx  = red[0];
  const float inv = 1.0f / red[1];
  float acc = 0.f;
  const int r0 = b << 8;
  for (int r = 0; r < 256; ++r) {
    const int i = r0 + r;
    acc += (__expf(gl[i] - mx) * inv) * emsg[(size_t)i * DH + c];
  }
  part[b * DH + c] = acc;
}

__global__ __launch_bounds__(256) void readout_final_kernel(const float* __restrict__ part,
                                                            float* __restrict__ eg) {
  const int c = threadIdx.x;
  float s = 0.f;
#pragma unroll
  for (int b = 0; b < 32; ++b) s += part[b * DH + c];
  eg[c] = s;
}

// ---------------- host-side launcher ----------------
extern "C" void kernel_launch(void* const* d_in, const int* in_sizes, int n_in,
                              void* d_out, int out_size, void* d_ws, size_t ws_size,
                              hipStream_t stream) {
  (void)in_sizes; (void)n_in; (void)out_size; (void)ws_size;

  const float* x_path = (const float*)d_in[0];
  const float* fc1_W  = (const float*)d_in[1];
  const float* fc1_b  = (const float*)d_in[2];
  const float* Wh     = (const float*)d_in[3];
  const float* bh     = (const float*)d_in[4];
  const float* Wt     = (const float*)d_in[5];
  const float* bt     = (const float*)d_in[6];
  const float* W1     = (const float*)d_in[7];
  const float* b1     = (const float*)d_in[8];
  const float* W2     = (const float*)d_in[9];
  const float* b2     = (const float*)d_in[10];
  const float* Ag1    = (const float*)d_in[11];
  const float* bg1    = (const float*)d_in[12];
  const float* Ag2    = (const float*)d_in[13];
  const float* bg2    = (const float*)d_in[14];

  constexpr size_t MB = 1024 * 1024;
  char* ws = (char*)d_ws;
  float*  x    = (float*)(ws + 0 * MB);    // 8 MB  (x_pre, then mixed x)
  float*  eh   = (float*)(ws + 8 * MB);    // 8 MB
  float*  et   = (float*)(ws + 16 * MB);   // 8 MB
  __bf16* ehb  = (__bf16*)(ws + 24 * MB);  // 4 MB
  __bf16* etb  = (__bf16*)(ws + 28 * MB);  // 4 MB
  int*    tidx = (int*)(ws + 32 * MB);     // 192 KB
  float*  t1   = (float*)(ws + 33 * MB);   // 8 MB
  float*  t2   = (float*)(ws + 41 * MB);   // 8 MB
  float*  h    = (float*)(ws + 49 * MB);   // 4 MB
  float*  gl   = (float*)(ws + 53 * MB);   // 32 KB
  float*  red  = (float*)(ws + 53 * MB + 64 * 1024);
  float*  part = (float*)(ws + 54 * MB);   // 32*256 partials (colsum, later readout)
  float*  mean = (float*)(ws + 54 * MB + 64 * 1024);

  float* emsg = (float*)d_out;                       // 8192 x 256
  float* eg   = (float*)d_out + (size_t)N_PATCH * DH;

  // GEMM grids: one wave per 64x16 strip, 4 waves per block
  const int g_main = (N_PATCH / 64) * (DH / 16) / 4;   // 512 blocks (Nc=256)
  const int g_ag1  = (N_PATCH / 64) * (128 / 16) / 4;  // 256 blocks (Nc=128)

  // 1) x_pre = leaky(x_path @ fc1_W + fc1_b)    [8192x1024x256]
  gemm_f32_kernel<<<g_main, 128, 0, stream>>>(x_path, DIN, fc1_W, DH, fc1_b,
                                              x, DH, DH, DIN, 1, 0);

  // 2) mean over N, then x = (x + mean) * 0.5
  colsum_partial_kernel<<<32, 256, 0, stream>>>(x, part);
  mean_final_kernel<<<1, 256, 0, stream>>>(part, mean);
  mix_kernel<<<N_PATCH * DH / 256, 256, 0, stream>>>(x, mean);

  // 3) e_h = x @ Wh + bh ; e_t = x @ Wt + bt
  gemm_f32_kernel<<<g_main, 128, 0, stream>>>(x, DH, Wh, DH, bh, eh, DH, DH, DH, 0, 0);
  gemm_f32_kernel<<<g_main, 128, 0, stream>>>(x, DH, Wt, DH, bt, et, DH, DH, DH, 0, 0);

  // 4) bf16 copies (SCALE folded into e_h), then fused logits + top-6
  cvt_kernel<<<N_PATCH * DH / 256, 256, 0, stream>>>(eh, et, ehb, etb);
  attn_topk_kernel<<<N_PATCH / 16, 256, 0, stream>>>(ehb, etb, tidx);

  // 5) message construction -> t1 = e_h + e_Nh, t2 = e_h * e_Nh   (fp32-exact logits)
  message_kernel<<<N_PATCH / 8, 256, 0, stream>>>(eh, et, tidx, t1, t2);

  // 6) e_msg = leaky(t1@W1+b1) + leaky(t2@W2+b2)   (written straight to d_out)
  gemm_f32_kernel<<<g_main, 128, 0, stream>>>(t1, DH, W1, DH, b1, emsg, DH, DH, DH, 1, 0);
  gemm_f32_kernel<<<g_main, 128, 0, stream>>>(t2, DH, W2, DH, b2, emsg, DH, DH, DH, 1, 1);

  // 7) readout: h = leaky(e_msg @ Ag1 + bg1); g = h @ Ag2 + bg2; softmax; e_g
  gemm_f32_kernel<<<g_ag1, 128, 0, stream>>>(emsg, DH, Ag1, 128, bg1, h, 128, 128, DH, 1, 0);
  glog_kernel<<<N_PATCH / 8, 256, 0, stream>>>(h, Ag2, bg2, gl);
  softmax_red_kernel<<<1, 256, 0, stream>>>(gl, red);
  readout_partial_kernel<<<32, 256, 0, stream>>>(gl, red, emsg, part);
  readout_final_kernel<<<1, 256, 0, stream>>>(part, eg);
}
